// GCN_simple_18107582120395
// MI455X (gfx1250) — compile-verified, matrix operand order checked by reading, not run on hardware
//
#include <hip/hip_runtime.h>
#include <hip/hip_bf16.h>

// ---------------------------------------------------------------------------
// GCN forward for MI455X (gfx1250, wave32, WMMA).
// Dense GEMMs -> v_wmma_f32_16x16x32_f16 (f16 in, f32 accumulate), with
// vectorized fragment loads (global_load_b128 for A, ds_load_b128 for B).
// Scatter/pool -> global_atomic_add_f32 via unsafeAtomicAdd.
// ---------------------------------------------------------------------------

#define NN   50000
#define EE   800000
#define GG   64
#define CIN  128
#define HH   128
#define LHH  256
#define NLAYERS 2
#define BN_EPS 1e-5f

typedef __attribute__((ext_vector_type(16))) _Float16 v16h;
typedef __attribute__((ext_vector_type(8)))  _Float16 v8h;
typedef __attribute__((ext_vector_type(8)))  float    v8f;

// ---------------------------------------------------------------------------
// Elementwise / utility kernels
// ---------------------------------------------------------------------------
__global__ void k_zero_f32(float* p, int n) {
    int i = blockIdx.x * blockDim.x + threadIdx.x;
    if (i < n) p[i] = 0.0f;
}

__global__ void k_cast_f32_to_f16(const float* __restrict__ a,
                                  _Float16* __restrict__ b, int n) {
    int i = blockIdx.x * blockDim.x + threadIdx.x;
    if (i < n) b[i] = (_Float16)a[i];
}

__global__ void k_deg_count(const int* __restrict__ dst, float* __restrict__ deg, int e) {
    int i = blockIdx.x * blockDim.x + threadIdx.x;
    if (i < e) unsafeAtomicAdd(&deg[dst[i]], 1.0f);
}

// deg -> deg^{-1/2} with self-loop (+1), in place
__global__ void k_deg_finalize(float* __restrict__ d, int n) {
    int i = blockIdx.x * blockDim.x + threadIdx.x;
    if (i < n) d[i] = rsqrtf(d[i] + 1.0f);
}

// ---------------------------------------------------------------------------
// Tiled WMMA GEMM: C[M,Nc] = A[M,K] @ B[K,Nc]; A,B f16 row-major, C f32.
// grid = (ceil(M/128), Nc/128), block = 256 (8 waves).
// Wave w owns rows [w*16, w*16+16) of the 128x128 block tile, all 8 col tiles.
//
// 16-bit fragment layout (ISA 7.12.2): lane (m16, g=lane/16) holds two
// contiguous runs of 8 K-values: K in [8g, 8g+8) then [16+8g, 16+8g+8),
// in vector-element order. So:
//   A frag  = two global_load_b128 from the lane's own row (row-major A).
//   B frag  = two ds_load_b128 from a column-major (transposed) LDS tile.
// B staging is hand-unrolled (exact trip count 2) to keep it straight-line:
// no exec-masked loop, uniform scalar branch around the prefetch.
// ---------------------------------------------------------------------------
__global__ __launch_bounds__(256)
void k_gemm_wmma_f16(const _Float16* __restrict__ A, const _Float16* __restrict__ B,
                     float* __restrict__ C, int M, int K, int Nc) {
    // BsT[col][k]: 128 cols x 32 K, row stride 40 f16 = 80 B (16B-aligned rows)
    __shared__ _Float16 BsT[128][40];

    const int tid  = threadIdx.x;
    const int wave = tid >> 5;
    const int lane = tid & 31;
    const int m16  = lane & 15;        // M (A/C) or N (B) index within 16x16 tile
    const int g    = lane >> 4;        // lane half: +8 to K (frags) / +8 to M (C)
    const int rowBlock = blockIdx.x * 128;
    const int colBlock = blockIdx.y * 128;

    // This lane's A row (clamped when OOB; result rows are store-guarded)
    const int arow  = rowBlock + wave * 16 + m16;
    const int arowC = (arow < M) ? arow : 0;
    const _Float16* Arow = A + (size_t)arowC * K;

    // Staging coordinates (chunk 0: rows 0-15, chunk 1: rows 16-31)
    const int sr  = tid >> 4;          // k row within tile (0..15)
    const int sc8 = (tid & 15) << 3;   // column group of 8

    v8f acc[8];
#pragma unroll
    for (int c = 0; c < 8; ++c) acc[c] = (v8f){0.f,0.f,0.f,0.f,0.f,0.f,0.f,0.f};

    for (int k0 = 0; k0 < K; k0 += 32) {
        // ---- stage B tile transposed: 2 b128 global reads, b16 LDS scatter
        const _Float16* src0 = B + (size_t)(k0 + sr)      * Nc + (colBlock + sc8);
        const _Float16* src1 = B + (size_t)(k0 + sr + 16) * Nc + (colBlock + sc8);
        v8h bv0 = *(const v8h*)src0;
        v8h bv1 = *(const v8h*)src1;
#pragma unroll
        for (int j = 0; j < 8; ++j) BsT[sc8 + j][sr]      = bv0[j];
#pragma unroll
        for (int j = 0; j < 8; ++j) BsT[sc8 + j][sr + 16] = bv1[j];
        if (k0 + 32 < K)               // uniform guard -> global_prefetch_b8
            __builtin_prefetch(src0 + (size_t)32 * Nc, 0, 3);
        __syncthreads();

        // ---- A fragment: two 16B loads straight from global
        v8h alo = *(const v8h*)(Arow + k0 + 8 * g);
        v8h ahi = *(const v8h*)(Arow + k0 + 16 + 8 * g);
        v16h afrag = __builtin_shufflevector(alo, ahi,
                         0,1,2,3,4,5,6,7, 8,9,10,11,12,13,14,15);

        // ---- 8 column tiles: two ds_load_b128 each + WMMA
#pragma unroll
        for (int c = 0; c < 8; ++c) {
            const _Float16* bp = &BsT[c * 16 + m16][0];
            v8h blo = *(const v8h*)(bp + 8 * g);
            v8h bhi = *(const v8h*)(bp + 16 + 8 * g);
            v16h bfrag = __builtin_shufflevector(blo, bhi,
                             0,1,2,3,4,5,6,7, 8,9,10,11,12,13,14,15);
            acc[c] = __builtin_amdgcn_wmma_f32_16x16x32_f16(
                false, afrag, false, bfrag, (short)0, acc[c], false, false);
        }
        __syncthreads();
    }

    // C/D layout: VGPR r, lanes 0-15 -> M=r, lanes 16-31 -> M=r+8; N = m16
    const int rowBase = rowBlock + wave * 16 + 8 * g;
#pragma unroll
    for (int c = 0; c < 8; ++c) {
        const int col = colBlock + c * 16 + m16;
#pragma unroll
        for (int r = 0; r < 8; ++r) {
            int row = rowBase + r;
            if (row < M) C[(size_t)row * Nc + col] = acc[c][r];
        }
    }
}

// ---------------------------------------------------------------------------
// Edge scatter: agg[dst,f] += hlin[src,f] * dis[src]*dis[dst]
// One thread per (edge, 4-feature chunk); float4 gather, 4 f32 atomics.
// ---------------------------------------------------------------------------
__global__ void k_edge_scatter(const float* __restrict__ hlin, const int* __restrict__ ei,
                               const float* __restrict__ dis, float* __restrict__ agg,
                               int e_cnt, int hc) {
    long long t = (long long)blockIdx.x * blockDim.x + threadIdx.x;
    const int perEdge = hc >> 2;
    if (t >= (long long)e_cnt * perEdge) return;
    int e = (int)(t / perEdge);
    int f = (int)(t % perEdge) << 2;
    int s = ei[e];
    int d = ei[e_cnt + e];
    float w = dis[s] * dis[d];
    const float4 hv = *(const float4*)(hlin + (size_t)s * hc + f);
    float* base = agg + (size_t)d * hc + f;
    unsafeAtomicAdd(base + 0, hv.x * w);
    unsafeAtomicAdd(base + 1, hv.y * w);
    unsafeAtomicAdd(base + 2, hv.z * w);
    unsafeAtomicAdd(base + 3, hv.w * w);
}

// relu(agg + hlin*snorm + b) -> f16 (input layer: no BN, feeds next GEMM)
__global__ void k_finalize_to_f16(const float* __restrict__ agg, const float* __restrict__ hlin,
                                  const float* __restrict__ dis, const float* __restrict__ bias,
                                  _Float16* __restrict__ out16, int n, int hc) {
    int i = blockIdx.x * blockDim.x + threadIdx.x;
    if (i >= n * hc) return;
    int nn = i / hc, f = i - nn * hc;
    float sn = dis[nn] * dis[nn];
    float v = agg[i] + hlin[i] * sn + bias[f];
    out16[i] = (_Float16)fmaxf(v, 0.0f);
}

// relu(agg + hlin*snorm + b) -> f32 (hidden layers: BN follows)
__global__ void k_finalize_to_f32(const float* __restrict__ agg, const float* __restrict__ hlin,
                                  const float* __restrict__ dis, const float* __restrict__ bias,
                                  float* __restrict__ out, int n, int hc) {
    int i = blockIdx.x * blockDim.x + threadIdx.x;
    if (i >= n * hc) return;
    int nn = i / hc, f = i - nn * hc;
    float sn = dis[nn] * dis[nn];
    float v = agg[i] + hlin[i] * sn + bias[f];
    out[i] = fmaxf(v, 0.0f);
}

// BatchNorm stats: one block per feature column
__global__ __launch_bounds__(256)
void k_bn_stats(const float* __restrict__ h, float* __restrict__ mean,
                float* __restrict__ rstd, int n, int hc) {
    const int f = blockIdx.x;
    float s = 0.0f, s2 = 0.0f;
    for (int r = threadIdx.x; r < n; r += 256) {
        float v = h[(size_t)r * hc + f];
        s += v; s2 += v * v;
    }
    __shared__ float sh[256], sh2[256];
    sh[threadIdx.x] = s; sh2[threadIdx.x] = s2;
    __syncthreads();
    for (int o = 128; o > 0; o >>= 1) {
        if (threadIdx.x < o) { sh[threadIdx.x] += sh[threadIdx.x + o];
                               sh2[threadIdx.x] += sh2[threadIdx.x + o]; }
        __syncthreads();
    }
    if (threadIdx.x == 0) {
        float m   = sh[0] / (float)n;
        float var = sh2[0] / (float)n - m * m;
        mean[f] = m;
        rstd[f] = rsqrtf(var + BN_EPS);
    }
}

// BN apply: emits f32 (reductions downstream) and f16 (next GEMM operand)
__global__ void k_bn_apply(const float* __restrict__ h, const float* __restrict__ mean,
                           const float* __restrict__ rstd, const float* __restrict__ gamma,
                           const float* __restrict__ beta, float* __restrict__ out32,
                           _Float16* __restrict__ out16, int n, int hc) {
    int i = blockIdx.x * blockDim.x + threadIdx.x;
    if (i >= n * hc) return;
    int f = i % hc;
    float v = (h[i] - mean[f]) * rstd[f] * gamma[f] + beta[f];
    out32[i] = v;
    out16[i] = (_Float16)v;
}

// global_add_pool over graphs
__global__ void k_pool_sum(const float* __restrict__ h, const int* __restrict__ batch,
                           float* __restrict__ pooled, int n, int hc) {
    int i = blockIdx.x * blockDim.x + threadIdx.x;
    if (i >= n * hc) return;
    int nn = i / hc, f = i - nn * hc;
    unsafeAtomicAdd(&pooled[(size_t)batch[nn] * hc + f], h[i]);
}

__global__ void k_bias_relu(const float* __restrict__ in, const float* __restrict__ b,
                            float* __restrict__ out, int m, int hc) {
    int i = blockIdx.x * blockDim.x + threadIdx.x;
    if (i >= m * hc) return;
    out[i] = fmaxf(in[i] + b[i % hc], 0.0f);
}

// out[g] = h[g,:] . Wo + bo
__global__ void k_final_dot(const float* __restrict__ h, const float* __restrict__ Wo,
                            const float* __restrict__ bo, float* __restrict__ out,
                            int gcnt, int kdim) {
    int g = blockIdx.x * blockDim.x + threadIdx.x;
    if (g >= gcnt) return;
    float s = bo[0];
    for (int k = 0; k < kdim; ++k) s += h[(size_t)g * kdim + k] * Wo[k];
    out[g] = s;
}

// ---------------------------------------------------------------------------
// Host side
// ---------------------------------------------------------------------------
static inline void* ws_bump(char*& p, size_t bytes) {
    char* r = p;
    p += (bytes + 255) & ~(size_t)255;
    return r;
}
static inline int ceil_div(int a, int b) { return (a + b - 1) / b; }

extern "C" void kernel_launch(void* const* d_in, const int* in_sizes, int n_in,
                              void* d_out, int out_size, void* d_ws, size_t ws_size,
                              hipStream_t stream) {
    (void)in_sizes; (void)n_in; (void)out_size; (void)ws_size;

    const float* x    = (const float*)d_in[0];
    const int*   ei   = (const int*)  d_in[1];   // [2,E]: src=ei[0..E), dst=ei[E..2E)
    const int*   batch= (const int*)  d_in[2];
    const float* Wi   = (const float*)d_in[3];
    const float* bi   = (const float*)d_in[4];
    const float* Ws   = (const float*)d_in[5];   // [L,H,H]
    const float* bs   = (const float*)d_in[6];   // [L,H]
    const float* gs   = (const float*)d_in[7];
    const float* bes  = (const float*)d_in[8];
    const float* W1   = (const float*)d_in[9];
    const float* b1   = (const float*)d_in[10];
    const float* g1   = (const float*)d_in[11];
    const float* be1  = (const float*)d_in[12];
    const float* W2   = (const float*)d_in[13];
    const float* b2   = (const float*)d_in[14];
    const float* g2   = (const float*)d_in[15];
    const float* be2  = (const float*)d_in[16];
    const float* Wo   = (const float*)d_in[17];
    const float* bo   = (const float*)d_in[18];
    float* out = (float*)d_out;

    // workspace carve-up (~90 MB)
    char* w = (char*)d_ws;
    float*    dis     = (float*)   ws_bump(w, (size_t)NN * 4);
    _Float16* hin16   = (_Float16*)ws_bump(w, (size_t)NN * HH * 2);
    float*    hlin    = (float*)   ws_bump(w, (size_t)NN * HH * 4);
    float*    agg     = (float*)   ws_bump(w, (size_t)NN * HH * 4);
    float*    hrelu   = (float*)   ws_bump(w, (size_t)NN * HH * 4);
    _Float16* Wi16    = (_Float16*)ws_bump(w, (size_t)CIN * HH * 2);
    _Float16* Ws16    = (_Float16*)ws_bump(w, (size_t)NLAYERS * HH * HH * 2);
    _Float16* W1h     = (_Float16*)ws_bump(w, (size_t)HH * LHH * 2);
    _Float16* W2h     = (_Float16*)ws_bump(w, (size_t)LHH * LHH * 2);
    float*    mean    = (float*)   ws_bump(w, (size_t)LHH * 4);
    float*    rstd    = (float*)   ws_bump(w, (size_t)LHH * 4);
    float*    pooled  = (float*)   ws_bump(w, (size_t)GG * HH * 4);
    _Float16* pooled16= (_Float16*)ws_bump(w, (size_t)GG * HH * 2);
    float*    t1      = (float*)   ws_bump(w, (size_t)GG * LHH * 4);
    _Float16* t1h     = (_Float16*)ws_bump(w, (size_t)GG * LHH * 2);
    float*    t2      = (float*)   ws_bump(w, (size_t)GG * LHH * 4);
    _Float16* t2h     = (_Float16*)ws_bump(w, (size_t)GG * LHH * 2);

    const int NH = NN * HH;
    const dim3 B256(256);

    // --- degree / symmetric normalization -------------------------------
    k_zero_f32<<<ceil_div(NN, 256), B256, 0, stream>>>(dis, NN);
    k_deg_count<<<ceil_div(EE, 256), B256, 0, stream>>>(ei + EE, dis, EE);
    k_deg_finalize<<<ceil_div(NN, 256), B256, 0, stream>>>(dis, NN);

    // --- f16 casts of activations and weights ---------------------------
    k_cast_f32_to_f16<<<ceil_div(NN * CIN, 256), B256, 0, stream>>>(x, hin16, NN * CIN);
    k_cast_f32_to_f16<<<ceil_div(CIN * HH, 256), B256, 0, stream>>>(Wi, Wi16, CIN * HH);
    k_cast_f32_to_f16<<<ceil_div(NLAYERS * HH * HH, 256), B256, 0, stream>>>(Ws, Ws16, NLAYERS * HH * HH);
    k_cast_f32_to_f16<<<ceil_div(HH * LHH, 256), B256, 0, stream>>>(W1, W1h, HH * LHH);
    k_cast_f32_to_f16<<<ceil_div(LHH * LHH, 256), B256, 0, stream>>>(W2, W2h, LHH * LHH);

    const dim3 gemmGridN(ceil_div(NN, 128), HH / 128);     // node-level GEMMs
    const long long scatterWork = (long long)EE * (HH / 4);
    const int scatterBlocks = (int)((scatterWork + 255) / 256);

    // --- input GCNConv + ReLU -------------------------------------------
    k_gemm_wmma_f16<<<gemmGridN, B256, 0, stream>>>(hin16, Wi16, hlin, NN, CIN, HH);
    k_zero_f32<<<ceil_div(NH, 256), B256, 0, stream>>>(agg, NH);
    k_edge_scatter<<<scatterBlocks, B256, 0, stream>>>(hlin, ei, dis, agg, EE, HH);
    k_finalize_to_f16<<<ceil_div(NH, 256), B256, 0, stream>>>(agg, hlin, dis, bi, hin16, NN, HH);

    // --- hidden GCNConv + ReLU + BatchNorm (L layers) -------------------
    for (int l = 0; l < NLAYERS; ++l) {
        k_gemm_wmma_f16<<<gemmGridN, B256, 0, stream>>>(hin16, Ws16 + (size_t)l * HH * HH,
                                                        hlin, NN, HH, HH);
        k_zero_f32<<<ceil_div(NH, 256), B256, 0, stream>>>(agg, NH);
        k_edge_scatter<<<scatterBlocks, B256, 0, stream>>>(hlin, ei, dis, agg, EE, HH);
        k_finalize_to_f32<<<ceil_div(NH, 256), B256, 0, stream>>>(agg, hlin, dis, bs + l * HH,
                                                                  hrelu, NN, HH);
        k_bn_stats<<<HH, B256, 0, stream>>>(hrelu, mean, rstd, NN, HH);
        // BN-applied f32 goes into `agg` (free now); f16 feeds the next GEMM
        k_bn_apply<<<ceil_div(NH, 256), B256, 0, stream>>>(hrelu, mean, rstd, gs + l * HH,
                                                           bes + l * HH, agg, hin16, NN, HH);
    }

    // --- global_add_pool -------------------------------------------------
    k_zero_f32<<<ceil_div(GG * HH, 256), B256, 0, stream>>>(pooled, GG * HH);
    k_pool_sum<<<ceil_div(NH, 256), B256, 0, stream>>>(agg, batch, pooled, NN, HH);
    k_cast_f32_to_f16<<<ceil_div(GG * HH, 256), B256, 0, stream>>>(pooled, pooled16, GG * HH);

    // --- MLP head ---------------------------------------------------------
    const dim3 headGrid1(1, LHH / 128);
    k_gemm_wmma_f16<<<headGrid1, B256, 0, stream>>>(pooled16, W1h, t1, GG, HH, LHH);
    k_bias_relu<<<ceil_div(GG * LHH, 256), B256, 0, stream>>>(t1, b1, t1, GG, LHH);
    k_bn_stats<<<LHH, B256, 0, stream>>>(t1, mean, rstd, GG, LHH);
    k_bn_apply<<<ceil_div(GG * LHH, 256), B256, 0, stream>>>(t1, mean, rstd, g1, be1,
                                                             t1, t1h, GG, LHH);

    k_gemm_wmma_f16<<<headGrid1, B256, 0, stream>>>(t1h, W2h, t2, GG, LHH, LHH);
    k_bias_relu<<<ceil_div(GG * LHH, 256), B256, 0, stream>>>(t2, b2, t2, GG, LHH);
    k_bn_stats<<<LHH, B256, 0, stream>>>(t2, mean, rstd, GG, LHH);
    k_bn_apply<<<ceil_div(GG * LHH, 256), B256, 0, stream>>>(t2, mean, rstd, g2, be2,
                                                             t2, t2h, GG, LHH);

    k_final_dot<<<1, GG, 0, stream>>>(t2, Wo, bo, out, GG, LHH);
}
